// SPIntraAttModuleV5_33346126086742
// MI455X (gfx1250) — matrix operand order repaired and checked by reading, not compiled, not run on hardware
//
#include <hip/hip_runtime.h>
#include <math.h>

typedef __attribute__((ext_vector_type(16))) _Float16 v16h;
typedef __attribute__((ext_vector_type(8)))  _Float16 v8h;
typedef __attribute__((ext_vector_type(8)))  float    v8f;
typedef __attribute__((ext_vector_type(4)))  float    v4f;
typedef __attribute__((ext_vector_type(4)))  unsigned int v4u;
typedef __attribute__((ext_vector_type(8)))  int      v8i;
typedef __attribute__((ext_vector_type(4)))  int      v4i;

#define BB 4
#define NN 4096
#define CC 256
#define SS 256
#define KK 32
#define NSAMP 30
#define EPSF 1e-8f
#define SCALE 0.0625f   // 1/sqrt(256)

// ---------------- LDS layout for attention kernel (bytes) ----------------
#define LDS_KEYS   0         // [32][256] f16  = 16384
#define LDS_KEYST  16384     // [256][32] f16  = 16384
#define LDS_X      32768     // [8][16][256] f16 = 65536
#define LDS_ATTN   98304     // [8][16][32] f16 = 8192
#define LDS_LOGW   106496    // [32] f32 = 128
#define LDS_WGT    106624    // [8][16] f32 = 512
#define LDS_TOTAL  107136

// ---------------- workspace layout (bytes) ----------------
#define WS_SIMS    0         // B*S*32 f32 = 128KB
#define WS_INDS    131072    // B*S*32 i32 = 128KB
#define WS_CNT     262144    // B*S*N  u8  = 4MB
#define WS_XH      4456448   // B*N*C  f16 = 8MB

// ============================ top-K kernel ============================
__global__ __launch_bounds__(256) void topk_kernel(const float* __restrict__ amat,
                                                   float* __restrict__ sims,
                                                   int* __restrict__ inds) {
    __shared__ float row[NN];
    __shared__ float rmax[256];
    __shared__ int   ridx[256];
    const int rowid = blockIdx.x;                    // b*S + s
    const float* src = amat + (size_t)rowid * NN;
    for (int i = threadIdx.x; i < NN; i += 256) row[i] = src[i];
    __syncthreads();
    for (int k = 0; k < KK; ++k) {
        float best = -__builtin_inff(); int bi = 0;
        for (int i = threadIdx.x; i < NN; i += 256) {
            float v = row[i];
            if (v > best) { best = v; bi = i; }
        }
        rmax[threadIdx.x] = best; ridx[threadIdx.x] = bi;
        __syncthreads();
        for (int off = 128; off > 0; off >>= 1) {
            if (threadIdx.x < off && rmax[threadIdx.x + off] > rmax[threadIdx.x]) {
                rmax[threadIdx.x] = rmax[threadIdx.x + off];
                ridx[threadIdx.x] = ridx[threadIdx.x + off];
            }
            __syncthreads();
        }
        if (threadIdx.x == 0) {
            sims[(size_t)rowid * KK + k] = rmax[0];
            inds[(size_t)rowid * KK + k] = ridx[0];
            row[ridx[0]] = -__builtin_inff();
        }
        __syncthreads();
    }
}

// ============================ zero kernel ============================
__global__ void zero_kernel(uint4* __restrict__ p, int n4) {
    int i = blockIdx.x * blockDim.x + threadIdx.x;
    if (i < n4) p[i] = make_uint4(0u, 0u, 0u, 0u);
}

// ==================== x f32 -> f16 pre-convert kernel ====================
__device__ __forceinline__ v8h cvt8(const float* __restrict__ p) {
    v4f f0 = *(const v4f*)p;
    v4f f1 = *(const v4f*)(p + 4);
    v8h h;
    h[0] = (_Float16)f0[0]; h[1] = (_Float16)f0[1];
    h[2] = (_Float16)f0[2]; h[3] = (_Float16)f0[3];
    h[4] = (_Float16)f1[0]; h[5] = (_Float16)f1[1];
    h[6] = (_Float16)f1[2]; h[7] = (_Float16)f1[3];
    return h;
}

__global__ __launch_bounds__(256) void cvt_kernel(const float* __restrict__ x,
                                                  _Float16* __restrict__ xh) {
    int i = blockIdx.x * 256 + threadIdx.x;          // 8 elements per thread
    *(v8h*)(xh + (size_t)i * 8) = cvt8(x + (size_t)i * 8);
}

// ============================ sampler kernel ============================
__device__ __forceinline__ unsigned pcg32(unsigned& st) {
    st = st * 747796405u + 2891336453u;
    unsigned w = ((st >> ((st >> 28u) + 4u)) ^ st) * 277803737u;
    return (w >> 22u) ^ w;
}

__global__ __launch_bounds__(256) void sample_kernel(const float* __restrict__ amat,
                                                     unsigned char* __restrict__ cnt) {
    int pix = blockIdx.x * 256 + threadIdx.x;        // b*N + n
    if (pix >= BB * NN) return;
    int b = pix >> 12, n = pix & (NN - 1);
    const float* col = amat + (size_t)b * SS * NN + n;
    float total = 0.f;
    for (int s = 0; s < SS; ++s) total += col[(size_t)s * NN];

    float us[NSAMP];
    unsigned st = (unsigned)pix * 0x9E3779B9u + 0x85EBCA6Bu;
    for (int t = 0; t < NSAMP; ++t) {
        unsigned r = pcg32(st);
        us[t] = ((float)(r >> 8) + 0.5f) * (1.f / 16777216.f) * total;
    }
    for (int i = 1; i < NSAMP; ++i) {                // insertion sort
        float v = us[i]; int j = i - 1;
        while (j >= 0 && us[j] > v) { us[j + 1] = us[j]; --j; }
        us[j + 1] = v;
    }
    unsigned char* crow = cnt + (size_t)b * SS * NN + n;
    float acc = 0.f; int p = 0;
    for (int s = 0; s < SS; ++s) {
        acc += col[(size_t)s * NN];
        int c = 0;
        while (p < NSAMP && us[p] <= acc) { ++c; ++p; }
        if (s == SS - 1) c += NSAMP - p;             // fp residue -> last bin
        if (c) crow[(size_t)s * NN] = (unsigned char)c;
    }
}

// ============================ WMMA attention kernel ============================
#define CAT16(lo, hi) __builtin_shufflevector(lo, hi, 0,1,2,3,4,5,6,7,8,9,10,11,12,13,14,15)

__global__ __launch_bounds__(256) void attn_kernel(const _Float16* __restrict__ xh,
                                                   const float* __restrict__ sims,
                                                   const int* __restrict__ inds,
                                                   const unsigned char* __restrict__ cnt,
                                                   float* __restrict__ out) {
    extern __shared__ char smem[];
    _Float16* keys  = (_Float16*)(smem + LDS_KEYS);    // [key][ch]
    _Float16* keysT = (_Float16*)(smem + LDS_KEYST);   // [ch][key]
    _Float16* Xt    = (_Float16*)(smem + LDS_X);       // [wave][pix][ch]
    _Float16* attnB = (_Float16*)(smem + LDS_ATTN);    // [wave][pix][key]
    float*    logw  = (float*)(smem + LDS_LOGW);
    float*    wgt   = (float*)(smem + LDS_WGT);

    const int b     = blockIdx.x >> 5;
    const int tg    = blockIdx.x & 31;
    const int tid   = threadIdx.x;
    const int wave  = tid >> 5;
    const int lane  = tid & 31;
    const int lhalf = lane >> 4;
    const int l16   = lane & 15;
    const int n0    = (tg * 8 + wave) * 16;            // this wave's pixel tile

    // ---- TDM: DMA this wave's X tile (16 pix x 256 ch f16 = 8KB, contiguous)
    // from global into LDS with a single tensor_load_to_lds.
    {
        unsigned long long ga =
            (unsigned long long)(uintptr_t)(xh + ((size_t)b * NN + n0) * CC);
        unsigned ga_lo = (unsigned)__builtin_amdgcn_readfirstlane((int)(ga & 0xFFFFFFFFull));
        unsigned ga_hi = (unsigned)__builtin_amdgcn_readfirstlane((int)(ga >> 32));
        unsigned lds   = (unsigned)__builtin_amdgcn_readfirstlane((int)(LDS_X + wave * 8192));
        v4u g0;
        g0[0] = 1u;                                   // count=1, user descriptor
        g0[1] = lds;                                  // lds_addr
        g0[2] = ga_lo;                                // global_addr[31:0]
        g0[3] = (ga_hi & 0x01FFFFFFu) | 0x80000000u;  // global_addr[56:32] | type=2
        v8i g1;
        g1[0] = 0x00030000;                           // data_size=3 (8B), no mask
        g1[1] = (int)(1024u << 16);                   // tensor_dim0 = 1024 (8B units)
        g1[2] = 0;                                    // tensor_dim0 hi, tensor_dim1=0
        g1[3] = (int)(1024u << 16);                   // tile_dim0 = 1024
        g1[4] = 0;                                    // tile_dim1/2 unused (1-D tile)
        g1[5] = 1024;                                 // tensor_dim0_stride = 1024
        g1[6] = 0;
        g1[7] = 0;
        v4i gz4 = {0, 0, 0, 0};
        v8i gz8 = {0, 0, 0, 0, 0, 0, 0, 0};
        __builtin_amdgcn_tensor_load_to_lds(g0, g1, gz4, gz4, gz8, 0);
        __builtin_amdgcn_s_wait_tensorcnt(0);
    }

    v8f acc[16];                                       // 16x256 f32 output tile
    #pragma unroll
    for (int i = 0; i < 16; ++i) {
        #pragma unroll
        for (int v = 0; v < 8; ++v) acc[i][v] = 0.f;
    }

    const _Float16* myX    = Xt + wave * 16 * CC;
    _Float16*       myAttn = attnB + wave * 16 * KK;
    float*          myW    = wgt + wave * 16;
    const int k0a = lhalf ? 8 : 0;                     // A-frag K half offset
    const int k0b = lhalf ? 16 : 0;                    // B-frag K half offset

    // ---- hoist s-invariant A fragments (X tile) into registers ----
    v16h aFrag[8];
    #pragma unroll
    for (int kc = 0; kc < 8; ++kc) {
        int cb = kc * 32;
        v8h alo = *(const v8h*)(myX + l16 * CC + cb + k0a);
        v8h ahi = *(const v8h*)(myX + l16 * CC + cb + k0a + 16);
        aFrag[kc] = CAT16(alo, ahi);
    }

    const unsigned long long xh_base = (unsigned long long)(uintptr_t)xh;

    for (int s = 0; s < SS; ++s) {
        __syncthreads();
        // ---- async gather of keys for superpixel s: per-lane indexed DMA ----
        // thread t: key k = t>>3, 32-ch chunk c0 = (t&7)*32; copies 64B via
        // 4x global_load_async_to_lds_b128 (ASYNCcnt-tracked, no VGPR staging)
        {
            int k  = tid >> 3;
            int c0 = (tid & 7) * 32;
            int gidx = inds[((size_t)b * SS + s) * KK + k];
            unsigned long long ga =
                xh_base + (((size_t)b * NN + gidx) * CC + c0) * 2ull;
            unsigned la = (unsigned)(LDS_KEYS + k * 512 + c0 * 2);
            #pragma unroll
            for (int j = 0; j < 64; j += 16) {
                unsigned long long gaj = ga + (unsigned)j;
                unsigned laj = la + (unsigned)j;
                asm volatile("global_load_async_to_lds_b128 %0, %1, off"
                             :: "v"(laj), "v"(gaj) : "memory");
            }
            if (tid < KK)
                logw[tid] = __logf(sims[((size_t)b * SS + s) * KK + tid] + EPSF);
            if (s + 1 < SS)
                __builtin_prefetch(&inds[((size_t)b * SS + s + 1) * KK], 0, 0);
            asm volatile("s_wait_asynccnt 0x0" ::: "memory");
            // transpose own chunk keys[k][c0..c0+31] -> keysT (same-thread data)
            #pragma unroll
            for (int j = 0; j < 32; j += 8) {
                v8h h = *(const v8h*)(keys + k * CC + c0 + j);
                #pragma unroll
                for (int e = 0; e < 8; ++e)
                    keysT[(c0 + j + e) * KK + k] = h[e];
            }
        }
        __syncthreads();

        // ---- per-pixel sample weights cnt/30; skip s if tile untouched ----
        unsigned char c8 = 0;
        if (lane < 16) {
            c8 = cnt[((size_t)b * SS + s) * NN + n0 + lane];
            myW[lane] = (float)c8 * (1.f / (float)NSAMP);
        }
        bool active = (__ballot(c8 != 0) != 0ull);
        asm volatile("s_wait_dscnt 0x0" ::: "memory");
        if (!active) continue;

        // ---- GEMM1: scores(16x32) = X(16x256) @ keys^T(256x32), f16 WMMA ----
        v8f c0s, c1s;
        #pragma unroll
        for (int v = 0; v < 8; ++v) { c0s[v] = 0.f; c1s[v] = 0.f; }
        #pragma unroll
        for (int kc = 0; kc < 8; ++kc) {
            int cb = kc * 32;
            v8h b0lo = *(const v8h*)(keys + l16 * CC + cb + k0a);
            v8h b0hi = *(const v8h*)(keys + l16 * CC + cb + k0a + 16);
            v16h b0 = CAT16(b0lo, b0hi);
            c0s = __builtin_amdgcn_wmma_f32_16x16x32_f16(false, aFrag[kc], false, b0,
                                                         (short)0, c0s, false, false);
            v8h b1lo = *(const v8h*)(keys + (16 + l16) * CC + cb + k0a);
            v8h b1hi = *(const v8h*)(keys + (16 + l16) * CC + cb + k0a + 16);
            v16h b1 = CAT16(b1lo, b1hi);
            c1s = __builtin_amdgcn_wmma_f32_16x16x32_f16(false, aFrag[kc], false, b1,
                                                         (short)0, c1s, false, false);
        }

        // ---- softmax over 32 keys (row m = v + 8*lhalf, col n = l16) ----
        float bias0 = logw[l16], bias1 = logw[l16 + 16];
        #pragma unroll
        for (int v = 0; v < 8; ++v) {
            float s0 = c0s[v] * SCALE + bias0;
            float s1 = c1s[v] * SCALE + bias1;
            float rm = fmaxf(s0, s1);
            #pragma unroll
            for (int off = 8; off >= 1; off >>= 1)
                rm = fmaxf(rm, __shfl_xor(rm, off, 32));
            float e0 = __expf(s0 - rm);
            float e1 = __expf(s1 - rm);
            float sum = e0 + e1;
            #pragma unroll
            for (int off = 8; off >= 1; off >>= 1)
                sum += __shfl_xor(sum, off, 32);
            float w = myW[v + 8 * lhalf] / sum;        // fold cnt/30 into attn
            myAttn[(v + 8 * lhalf) * KK + l16]      = (_Float16)(e0 * w);
            myAttn[(v + 8 * lhalf) * KK + l16 + 16] = (_Float16)(e1 * w);
        }
        asm volatile("s_wait_dscnt 0x0" ::: "memory");

        // ---- GEMM2: out(16x256) += attn(16x32) @ keys(32x256) ----
        v8h palo = *(const v8h*)(myAttn + l16 * KK + k0a);
        v8h pahi = *(const v8h*)(myAttn + l16 * KK + k0a + 16);
        v16h pa = CAT16(palo, pahi);
        #pragma unroll
        for (int cbk = 0; cbk < 16; ++cbk) {
            const _Float16* tcol = keysT + (cbk * 16 + l16) * KK;
            v8h blo = *(const v8h*)(tcol + k0a);
            v8h bhi = *(const v8h*)(tcol + k0a + 16);
            v16h bb = CAT16(blo, bhi);
            acc[cbk] = __builtin_amdgcn_wmma_f32_16x16x32_f16(false, pa, false, bb,
                                                              (short)0, acc[cbk], false, false);
        }
    }

    // ---- store output tile (each wave owns its 16 pixels exclusively) ----
    float* ob = out + ((size_t)b * NN + n0) * CC;
    #pragma unroll
    for (int cbk = 0; cbk < 16; ++cbk) {
        #pragma unroll
        for (int v = 0; v < 8; ++v)
            ob[(v + 8 * lhalf) * CC + cbk * 16 + l16] = acc[cbk][v];
    }
}

// ============================ launcher ============================
extern "C" void kernel_launch(void* const* d_in, const int* in_sizes, int n_in,
                              void* d_out, int out_size, void* d_ws, size_t ws_size,
                              hipStream_t stream) {
    const float* x    = (const float*)d_in[0];
    const float* amat = (const float*)d_in[1];
    (void)in_sizes; (void)n_in; (void)out_size; (void)ws_size;
    float* out = (float*)d_out;

    char* ws = (char*)d_ws;
    float*         sims = (float*)(ws + WS_SIMS);
    int*           inds = (int*)(ws + WS_INDS);
    unsigned char* cnt  = (unsigned char*)(ws + WS_CNT);
    _Float16*      xh   = (_Float16*)(ws + WS_XH);

    hipLaunchKernelGGL(topk_kernel, dim3(BB * SS), dim3(256), 0, stream,
                       amat, sims, inds);
    int n4 = (BB * SS * NN) / 16;
    hipLaunchKernelGGL(zero_kernel, dim3((n4 + 255) / 256), dim3(256), 0, stream,
                       (uint4*)cnt, n4);
    hipLaunchKernelGGL(cvt_kernel, dim3((BB * NN * CC / 8) / 256), dim3(256), 0, stream,
                       x, xh);
    hipLaunchKernelGGL(sample_kernel, dim3((BB * NN) / 256), dim3(256), 0, stream,
                       amat, cnt);
    hipLaunchKernelGGL(attn_kernel, dim3(BB * 32), dim3(256), LDS_TOTAL, stream,
                       xh, sims, inds, cnt, out);
}
